// WAttentionNet_19026705122020
// MI455X (gfx1250) — matrix operand order repaired and checked by reading, not compiled, not run on hardware
//
#include <hip/hip_runtime.h>
#include <hip/hip_bf16.h>
#include <math.h>

typedef __attribute__((ext_vector_type(16))) _Float16 v16h;
typedef __attribute__((ext_vector_type(8)))  _Float16 v8h;
typedef __attribute__((ext_vector_type(8)))  float    v8f;

#define MODE_STORE 0
#define MODE_ADD   1
#define MODE_SILU  2

#define NW 2            // waves per workgroup (64 threads)

// LDS arena offsets (in halfs). Regions are reused across phases:
//  self-attention phase: CTX, BUF1(xn->O), KH, VH, QH
//  FF phase:             CTX, BUF1(xn2), FFMID (over KH/VH/QH tail)
//  qg/cross/nu phases:   CTX, BUF1(kvn), KH, VH, TOK (over dead QH), Q, QN
#define OCTX   0        // 7*16*128
#define OBUF1  14336    // 7*16*128 (xn / O / kvn)
#define OKH    28672    // 7*16*128
#define OVH    43008    // 7*16*128
#define OFFMID 28672    // up to 4*16*512 (FF phase only)
#define OQH    57344    // 7*16*128 (self-attention phase only)
#define OTOK   57344    // qg staging / cross qh / per-wave nu scratch
#define OQ     65536    // 2*16*128
#define OQN    69632    // 2*16*128
#define STOT   73728

template<typename WT>
struct BlockW {
  const float *ln1_g,*ln1_b;
  const WT *wq_w; const float *wq_b;
  const WT *wk_w; const float *wk_b;
  const WT *wv_w; const float *wv_b;
  const WT *wo_w; const float *wo_b;
  const float *ln2_g,*ln2_b;
  const WT *ff1_w; const float *ff1_b;
  const WT *ff2_w; const float *ff2_b;
  const float *lnkv_g,*lnkv_b;
};
template<typename WT>
struct NetW {
  const float *x;
  const float *std_mean,*std_scale;
  const float *lep_w,*lep_b,*jet_w,*jet_b,*met_w,*met_b,*hl_w,*hl_b;
  const float *qgln_g,*qgln_b,*qg1_w,*qg1_b;
  const WT *qg2_w; const float *qg2_b;
  BlockW<WT> sb[3];
  BlockW<WT> cb[3];
  const float *nuln_g,*nuln_b;
  const WT *nu1_w; const float *nu1_b;
  const float *nu2_w,*nu2_b;
};

// B-fragment loader: f32 weights (convert inline) or pre-converted f16 (raw 32B load)
__device__ __forceinline__ v16h loadBfrag(const float* wp) {
  v16h bb;
#pragma unroll
  for (int i = 0; i < 16; ++i) bb[i] = (_Float16)wp[i];
  return bb;
}
__device__ __forceinline__ v16h loadBfrag(const _Float16* wp) {
  return *(const v16h*)wp;
}

// Y[16xN] (ldy, per-mtile stride ystride) = X[16xK] @ W[N,K]^T + bias, WMMA f16->f32.
// B fragments cached in registers across the mtiles loop (token reuse).
// N-tile loop strided (n0start, n0step) so multiple waves can split one GEMM.
template<int K, int MODE, typename WT>
__device__ __forceinline__ void gemm16(
    _Float16* __restrict__ Y, int ldy, int ystride,
    const _Float16* __restrict__ X, int xstride, int mtiles,
    const WT* __restrict__ W, const float* __restrict__ bias,
    int N, int n0start, int n0step, int lane)
{
  constexpr int NK = K / 32;
  const int c16 = lane & 15;
  const int hi  = lane >> 4;
  for (int n0 = n0start; n0 < N; n0 += n0step) {
    v16h bf[NK];
    const WT* wrow = W + (size_t)(n0 + c16) * K + hi * 16;
#pragma unroll
    for (int kt = 0; kt < NK; ++kt) bf[kt] = loadBfrag(wrow + kt * 32);
    const float bv = bias[n0 + c16];
    for (int mt = 0; mt < mtiles; ++mt) {
      const _Float16* Xr = X + mt * xstride + c16 * K + hi * 16;
      v8f acc = {bv, bv, bv, bv, bv, bv, bv, bv};
#pragma unroll
      for (int kt = 0; kt < NK; ++kt) {
        const v8h* xp = (const v8h*)(Xr + kt * 32);
        v8h lo = xp[0];
        v8h hh = xp[1];   // +8 halfs
        v16h av;
#pragma unroll
        for (int i = 0; i < 8; ++i) { av[i] = lo[i]; av[8 + i] = hh[i]; }
        acc = __builtin_amdgcn_wmma_f32_16x16x32_f16(
            false, av, false, bf[kt], (short)0, acc, false, false);
      }
      _Float16* Yp = Y + mt * ystride + (hi * 8) * ldy + n0 + c16;
#pragma unroll
      for (int j = 0; j < 8; ++j) {
        float v = acc[j];
        _Float16* yp = Yp + j * ldy;
        if (MODE == MODE_ADD)       *yp = (_Float16)((float)*yp + v);
        else if (MODE == MODE_SILU) *yp = (_Float16)(v / (1.f + __expf(-v)));
        else                        *yp = (_Float16)v;
      }
    }
  }
}

// LayerNorm over 128 channels for 16 rows; 2 lanes per row, paired via shfl_xor(16).
__device__ __forceinline__ void ln16(
    _Float16* __restrict__ dst, const _Float16* __restrict__ src,
    const float* __restrict__ g, const float* __restrict__ b,
    int lane, bool do_silu)
{
  const int r = lane & 15, hi = lane >> 4;
  const _Float16* sp = src + r * 128 + hi * 64;
  float s = 0.f, s2 = 0.f;
#pragma unroll
  for (int c = 0; c < 64; ++c) { float v = (float)sp[c]; s += v; s2 += v * v; }
  s  += __shfl_xor(s, 16);
  s2 += __shfl_xor(s2, 16);
  const float m  = s * (1.f / 128.f);
  const float rs = rsqrtf(s2 * (1.f / 128.f) - m * m + 1e-5f);
  _Float16* dp = dst + r * 128 + hi * 64;
  const float* gp = g + hi * 64;
  const float* bp = b + hi * 64;
#pragma unroll
  for (int c = 0; c < 64; ++c) {
    float v = ((float)sp[c] - m) * rs * gp[c] + bp[c];
    if (do_silu) v = v / (1.f + __expf(-v));
    dp[c] = (_Float16)v;
  }
}

// 8-head attention for one query token over 7 keys; lane handles (row, 4 heads).
__device__ __forceinline__ void attend(
    _Float16* __restrict__ OT, const _Float16* __restrict__ QH,
    const _Float16* __restrict__ KHp, const _Float16* __restrict__ VHp,
    const int* __restrict__ maskb, int lane)
{
  const int r = lane & 15, hb = (lane >> 4) * 4;
  const int mk = maskb[r];
  for (int hh = 0; hh < 4; ++hh) {
    const int h = hb + hh;
    float qv[16];
#pragma unroll
    for (int d = 0; d < 16; ++d) qv[d] = (float)QH[r * 128 + h * 16 + d];
    float s[7]; float mx = -3.0e38f;
#pragma unroll
    for (int k = 0; k < 7; ++k) {
      const _Float16* kp = KHp + k * 2048 + r * 128 + h * 16;
      float dot = 0.f;
#pragma unroll
      for (int d = 0; d < 16; ++d) dot += qv[d] * (float)kp[d];
      dot *= 0.25f;                         // 1/sqrt(16)
      if (mk & (1 << k)) dot = -1.0e9f;     // key mask
      s[k] = dot; mx = fmaxf(mx, dot);
    }
    float den = 0.f;
#pragma unroll
    for (int k = 0; k < 7; ++k) { float e = __expf(s[k] - mx); s[k] = e; den += e; }
    const float rd = 1.f / den;
#pragma unroll
    for (int d = 0; d < 16; ++d) {
      float o = 0.f;
#pragma unroll
      for (int k = 0; k < 7; ++k) o += s[k] * (float)VHp[k * 2048 + r * 128 + h * 16 + d];
      OT[r * 128 + h * 16 + d] = (_Float16)(o * rd);
    }
  }
}

// Small-K embedding: dst[16][128] = xs[:,xoff:xoff+K] @ W^T + b (scalar, tiny)
__device__ __forceinline__ void embed16(
    _Float16* __restrict__ dst, const float* __restrict__ xs, int xoff,
    const float* __restrict__ W, const float* __restrict__ b, int K, int tid)
{
  for (int idx = tid; idx < 2048; idx += 32 * NW) {
    const int r = idx >> 7, c = idx & 127;
    float acc = b[c];
    const float* xr = xs + r * 32 + xoff;
    const float* wr = W + c * K;
    for (int k = 0; k < K; ++k) acc += xr[k] * wr[k];
    dst[r * 128 + c] = (_Float16)acc;
  }
}

template<typename WT>
__global__ __launch_bounds__(32 * NW)
void wnet_kernel(NetW<WT> P, float* __restrict__ out, int Btot)
{
  __shared__ alignas(16) _Float16 S[STOT];
  __shared__ alignas(16) float XR[512];
  __shared__ alignas(16) float XS[512];
  __shared__ float QIN[160];
  __shared__ float NUOUT[96];
  __shared__ int   MASKB[16];

  const int tid  = threadIdx.x;
  const int lane = tid & 31;
  const int wid  = tid >> 5;          // wave id in {0,1}
  const int wg   = blockIdx.x;
  _Float16* TOKw = S + OTOK + wid * 4096;   // per-wave scratch (qg/nu phases)

  // ---- load 16-event tile, standardize ----
  {
    const float ms = P.std_mean[lane], sc = P.std_scale[lane];
    for (int i = wid; i < 16; i += NW) {
      float v = P.x[((size_t)wg * 16 + i) * 32 + lane];
      XR[i * 32 + lane] = v;
      XS[i * 32 + lane] = (v - ms) / sc;
    }
  }
  __syncthreads();
  if (tid < 16) {
    const float* xr = XR + tid * 32;
    float s2 = 0, s3 = 0, s4 = 0;
    for (int j = 0; j < 4; ++j) { s2 += fabsf(xr[8+j]); s3 += fabsf(xr[12+j]); s4 += fabsf(xr[16+j]); }
    MASKB[tid] = ((s2==0.f)?4:0) | ((s3==0.f)?8:0) | ((s4==0.f)?16:0);
  }

  // ---- embeddings -> context tokens ----
  embed16(S+OCTX+0*2048, XS,  0, P.lep_w, P.lep_b,  4, tid);
  embed16(S+OCTX+1*2048, XS,  4, P.lep_w, P.lep_b,  4, tid);
  embed16(S+OCTX+2*2048, XS,  8, P.jet_w, P.jet_b,  4, tid);
  embed16(S+OCTX+3*2048, XS, 12, P.jet_w, P.jet_b,  4, tid);
  embed16(S+OCTX+4*2048, XS, 16, P.jet_w, P.jet_b,  4, tid);
  embed16(S+OCTX+5*2048, XS, 20, P.met_w, P.met_b,  2, tid);
  embed16(S+OCTX+6*2048, XS, 22, P.hl_w,  P.hl_b,  10, tid);
  __syncthreads();

  // ---- 3 self-attention blocks ----
  for (int blk = 0; blk < 3; ++blk) {
    const BlockW<WT>& p = P.sb[blk];
    for (int t = wid; t < 7; t += NW) ln16(S+OBUF1+t*2048, S+OCTX+t*2048, p.ln1_g, p.ln1_b, lane, false);
    __syncthreads();
    // Q/K/V projections, all 7 tokens batched, N-tiles split across waves (B frags cached)
    gemm16<128, MODE_STORE>(S+OKH, 128, 2048, S+OBUF1, 2048, 7, p.wk_w, p.wk_b, 128, wid*16, 32, lane);
    gemm16<128, MODE_STORE>(S+OVH, 128, 2048, S+OBUF1, 2048, 7, p.wv_w, p.wv_b, 128, wid*16, 32, lane);
    gemm16<128, MODE_STORE>(S+OQH, 128, 2048, S+OBUF1, 2048, 7, p.wq_w, p.wq_b, 128, wid*16, 32, lane);
    __syncthreads();
    // scalar attention (tokens split across waves); O overwrites dead xn in BUF1
    for (int t = wid; t < 7; t += NW)
      attend(S+OBUF1+t*2048, S+OQH+t*2048, S+OKH, S+OVH, MASKB, lane);
    __syncthreads();
    // output projection, all tokens batched: CTX += O @ Wo^T + b
    gemm16<128, MODE_ADD>(S+OCTX, 128, 2048, S+OBUF1, 2048, 7, p.wo_w, p.wo_b, 128, wid*16, 32, lane);
    __syncthreads();
    for (int t = wid; t < 7; t += NW) ln16(S+OBUF1+t*2048, S+OCTX+t*2048, p.ln2_g, p.ln2_b, lane, false);
    __syncthreads();
    // FF in two token groups (FFMID overlaps KH/VH/QH, dead after attention); N split across waves
    gemm16<128, MODE_SILU>(S+OFFMID, 512, 8192, S+OBUF1, 2048, 4, p.ff1_w, p.ff1_b, 512, wid*16, 32, lane);
    __syncthreads();
    gemm16<512, MODE_ADD >(S+OCTX, 128, 2048, S+OFFMID, 8192, 4, p.ff2_w, p.ff2_b, 128, wid*16, 32, lane);
    __syncthreads();
    gemm16<128, MODE_SILU>(S+OFFMID, 512, 8192, S+OBUF1+4*2048, 2048, 3, p.ff1_w, p.ff1_b, 512, wid*16, 32, lane);
    __syncthreads();
    gemm16<512, MODE_ADD >(S+OCTX+4*2048, 128, 2048, S+OFFMID, 8192, 3, p.ff2_w, p.ff2_b, 128, wid*16, 32, lane);
    __syncthreads();
  }

  // ---- query generator: LN(10) -> l1+silu (scalar) -> l2 (WMMA) ----
  if (tid < 16) {
    const int r = tid;
    float v[10];
    for (int j = 0; j < 8; ++j) v[j] = XS[r*32 + j];
    v[8] = XS[r*32 + 20]; v[9] = XS[r*32 + 21];
    float m = 0; for (int j = 0; j < 10; ++j) m += v[j]; m *= 0.1f;
    float var = 0; for (int j = 0; j < 10; ++j) { float d = v[j]-m; var += d*d; } var *= 0.1f;
    const float rs = rsqrtf(var + 1e-5f);
    for (int j = 0; j < 10; ++j) QIN[r*10 + j] = (v[j]-m)*rs*P.qgln_g[j] + P.qgln_b[j];
  }
  __syncthreads();
  for (int idx = tid; idx < 2048; idx += 32 * NW) {
    const int r = idx >> 7, c = idx & 127;
    float acc = P.qg1_b[c];
    for (int k = 0; k < 10; ++k) acc += QIN[r*10 + k] * P.qg1_w[c*10 + k];
    acc = acc / (1.f + __expf(-acc));
    S[OTOK + r*128 + c] = (_Float16)acc;
  }
  __syncthreads();
  // qg_l2 -> two query tokens: one per wave
  gemm16<128, MODE_STORE>(S+OQ+wid*2048, 128, 0, S+OTOK, 0, 1,
                          P.qg2_w + (size_t)wid*128*128, P.qg2_b + wid*128, 128, 0, 16, lane);
  __syncthreads();

  // ---- 3 cross-attention blocks ----
  for (int blk = 0; blk < 3; ++blk) {
    const BlockW<WT>& p = P.cb[blk];
    ln16(S+OQN+wid*2048, S+OQ+wid*2048, p.ln1_g, p.ln1_b, lane, false);
    for (int t = wid; t < 7; t += NW) ln16(S+OBUF1+t*2048, S+OCTX+t*2048, p.lnkv_g, p.lnkv_b, lane, false);
    __syncthreads();
    gemm16<128, MODE_STORE>(S+OKH, 128, 2048, S+OBUF1, 2048, 7, p.wk_w, p.wk_b, 128, wid*16, 32, lane);
    gemm16<128, MODE_STORE>(S+OVH, 128, 2048, S+OBUF1, 2048, 7, p.wv_w, p.wv_b, 128, wid*16, 32, lane);
    gemm16<128, MODE_STORE>(S+OTOK, 128, 2048, S+OQN, 2048, 2, p.wq_w, p.wq_b, 128, wid*16, 32, lane);
    __syncthreads();
    { // scalar attention, one query token per wave; O overwrites dead qn in QN
      const int t = wid;
      attend(S+OQN+t*2048, S+OTOK+t*2048, S+OKH, S+OVH, MASKB, lane);
    }
    __syncthreads();
    gemm16<128, MODE_ADD>(S+OQ, 128, 2048, S+OQN, 2048, 2, p.wo_w, p.wo_b, 128, wid*16, 32, lane);
    __syncthreads();
    ln16(S+OQN+wid*2048, S+OQ+wid*2048, p.ln2_g, p.ln2_b, lane, false);
    __syncthreads();
    gemm16<128, MODE_SILU>(S+OFFMID, 512, 8192, S+OQN, 2048, 2, p.ff1_w, p.ff1_b, 512, wid*16, 32, lane);
    __syncthreads();
    gemm16<512, MODE_ADD >(S+OQ, 128, 2048, S+OFFMID, 8192, 2, p.ff2_w, p.ff2_b, 128, wid*16, 32, lane);
    __syncthreads();
  }

  // ---- nu head: one token per wave ----
  {
    const int t = wid;
    ln16(TOKw, S+OQ+t*2048, P.nuln_g, P.nuln_b, lane, true);   // silu(LN(.))
    gemm16<128, MODE_SILU>(TOKw+2048, 32, 0, TOKw, 0, 1, P.nu1_w, P.nu1_b, 32, 0, 16, lane);
    for (int idx = lane; idx < 48; idx += 32) {
      const int r = idx / 3, c = idx - r * 3;
      float acc = P.nu2_b[c];
      const _Float16* hp = TOKw + 2048 + r * 32;
      const float* wr = P.nu2_w + c * 32;
      for (int k = 0; k < 32; ++k) acc += (float)hp[k] * wr[k];
      NUOUT[t*48 + r*3 + c] = acc;
    }
  }
  __syncthreads();

  // ---- assemble w / sw outputs ----
  if (tid < 16) {
    const int r = tid;
    const size_t row = (size_t)wg * 16 + r;
    const float* a0 = NUOUT + r*3;
    const float* a1 = NUOUT + 48 + r*3;
    const float e0 = sqrtf(a0[0]*a0[0] + a0[1]*a0[1] + a0[2]*a0[2]);
    const float e1 = sqrtf(a1[0]*a1[0] + a1[1]*a1[1] + a1[2]*a1[2]);
    const float* xr = XR + r*32;
    float* w  = out + row * 8;
    float* sw = out + (size_t)Btot * 8 + row * 8;
    w[0]=xr[0]+a0[0]; w[1]=xr[1]+a0[1]; w[2]=xr[2]+a0[2]; w[3]=xr[3]+e0;
    w[4]=xr[4]+a1[0]; w[5]=xr[5]+a1[1]; w[6]=xr[6]+a1[2]; w[7]=xr[7]+e1;
    sw[0]=xr[0]+a1[0]; sw[1]=xr[1]+a1[1]; sw[2]=xr[2]+a1[2]; sw[3]=xr[3]+e1;
    sw[4]=xr[4]+a0[0]; sw[5]=xr[5]+a0[1]; sw[6]=xr[6]+a0[2]; sw[7]=xr[7]+e0;
  }
}

// f32 -> f16 weight pre-conversion (amortized over all 8192 workgroups)
__global__ void cvt_kernel(const float* __restrict__ src, _Float16* __restrict__ dst, int n)
{
  const int i = blockIdx.x * 256 + threadIdx.x;
  if (i < n) dst[i] = (_Float16)src[i];
}

// -------- host side --------
// setup_inputs() insertion order: x, std_mean, std_scale, lep{w,b}, jet{w,b},
// met{w,b}, hl{w,b}, qg_ln{g,b}, qg_l1{w,b}, qg_l2{w,b},
// self_blocks[3]{ln1{g,b},wq{w,b},wk{w,b},wv{w,b},wo{w,b},ln2{g,b},ff1{w,b},ff2{w,b}},
// cross_blocks[3]{... + ln_kv{g,b}}, nu_ln{g,b}, nu_l1{w,b}, nu_l2{w,b}

template<typename WT, typename PLACE>
static void buildNet(NetW<WT>& P, void* const* d_in, PLACE place)
{
  auto F = [&](int i) { return (const float*)d_in[i]; };
  P.x        = F(0);
  P.std_mean = F(1);  P.std_scale = F(2);
  P.lep_w = F(3);  P.lep_b = F(4);
  P.jet_w = F(5);  P.jet_b = F(6);
  P.met_w = F(7);  P.met_b = F(8);
  P.hl_w  = F(9);  P.hl_b  = F(10);
  P.qgln_g = F(11); P.qgln_b = F(12);
  P.qg1_w  = F(13); P.qg1_b  = F(14);
  P.qg2_w  = place(F(15), 256 * 128); P.qg2_b = F(16);
  int idx = 17;
  for (int pass = 0; pass < 2; ++pass) {
    for (int i = 0; i < 3; ++i) {
      BlockW<WT>& b = (pass == 0) ? P.sb[i] : P.cb[i];
      b.ln1_g = F(idx+0);  b.ln1_b = F(idx+1);
      b.wq_w  = place(F(idx+2),  128*128); b.wq_b  = F(idx+3);
      b.wk_w  = place(F(idx+4),  128*128); b.wk_b  = F(idx+5);
      b.wv_w  = place(F(idx+6),  128*128); b.wv_b  = F(idx+7);
      b.wo_w  = place(F(idx+8),  128*128); b.wo_b  = F(idx+9);
      b.ln2_g = F(idx+10); b.ln2_b = F(idx+11);
      b.ff1_w = place(F(idx+12), 512*128); b.ff1_b = F(idx+13);
      b.ff2_w = place(F(idx+14), 128*512); b.ff2_b = F(idx+15);
      if (pass == 1) { b.lnkv_g = F(idx+16); b.lnkv_b = F(idx+17); idx += 18; }
      else           { b.lnkv_g = nullptr;   b.lnkv_b = nullptr;   idx += 16; }
    }
  }
  P.nuln_g = F(idx++); P.nuln_b = F(idx++);
  P.nu1_w  = place(F(idx++), 32*128); P.nu1_b = F(idx++);
  P.nu2_w  = F(idx++); P.nu2_b = F(idx++);
}

extern "C" void kernel_launch(void* const* d_in, const int* in_sizes, int n_in,
                              void* d_out, int out_size, void* d_ws, size_t ws_size,
                              hipStream_t stream)
{
  (void)n_in; (void)out_size;
  const int B = in_sizes[0] / 32;   // 131072
  const dim3 grid(B / 16), block(32 * NW);

  // total f16 weight elements: 6*(4*16384 + 2*65536) + 32768 + 4096 = 1,216,512
  const size_t needed = 1216512 * sizeof(_Float16);
  if (ws_size >= needed) {
    _Float16* wbase = (_Float16*)d_ws;
    size_t off = 0;
    auto place = [&](const float* src, int n) -> const _Float16* {
      _Float16* dst = wbase + off;
      cvt_kernel<<<dim3((n + 255) / 256), dim3(256), 0, stream>>>(src, dst, n);
      off += (size_t)n;
      return dst;
    };
    NetW<_Float16> P;
    buildNet<_Float16>(P, d_in, place);
    wnet_kernel<_Float16><<<grid, block, 0, stream>>>(P, (float*)d_out, B);
  } else {
    auto place = [&](const float* src, int n) -> const float* { (void)n; return src; };
    NetW<float> P;
    buildNet<float>(P, d_in, place);
    wnet_kernel<float><<<grid, block, 0, stream>>>(P, (float*)d_out, B);
  }
}
